// CondRQsplineSeparatedAndCond2dToy_51101520887947
// MI455X (gfx1250) — matrix operand order repaired and checked by reading, not compiled, not run on hardware
//
#include <hip/hip_runtime.h>

typedef __attribute__((ext_vector_type(16))) _Float16 v16h;
typedef __attribute__((ext_vector_type(8)))  _Float16 v8h;
typedef __attribute__((ext_vector_type(8)))  float    v8f;
typedef __attribute__((ext_vector_type(4)))  int      v4i;

// ---------------- constants ----------------
#define B_IMG   16
#define H_IMG   256
#define W_IMG   256
#define PLANE   (H_IMG * W_IMG)          // 65536
#define IMG_STRIDE (8 * PLANE)           // 8 channels per image
#define HID     64
#define TILE    16
#define HT      18                        // h tile (halo 1)
#define IT      20                        // input tile (halo 2)
#define NP      (HT * HT)                 // 324 valid h pixels
#define MROWS   336                       // padded to 21 M-tiles
#define MT      21

// f16 weight workspace layout (halfs): Bt0 | Bt1..Bt4 | Bz2(9 x [16][64])
#define W_B0_OFF   0
#define W_B1_OFF   4096
#define W_BZ_OFF   20480
#define W_TOTAL_H  29696                  // halfs
#define W_TOTAL_B  (W_TOTAL_H * 2)        // 59392 bytes
#define W_TOTAL_B128 (W_TOTAL_B / 16)     // 3712 x 16B transfers

// LDS layout (bytes): sW(59392) | sHA(43008) | sHB(43008) | sFeat(16384 f32, aliased by sIn) | sRed(1024)
#define LDS_BYTES 162816

#define AS1 __attribute__((address_space(1)))
#define AS3 __attribute__((address_space(3)))

// ---------------- CDNA5 async global->LDS copy (ASYNCcnt path) ----------------
__device__ __forceinline__ void async_g2l_b128(unsigned int lds_off, const void* g) {
#if __has_builtin(__builtin_amdgcn_global_load_async_to_lds_b128)
  __builtin_amdgcn_global_load_async_to_lds_b128(
      (AS1 v4i*)(uintptr_t)g, (AS3 v4i*)(uintptr_t)lds_off, 0, 0);
#else
  asm volatile("global_load_async_to_lds_b128 %0, %1, off"
               :: "v"(lds_off), "v"((unsigned long long)(uintptr_t)g) : "memory");
#endif
}
__device__ __forceinline__ void wait_asynccnt0() {
#if __has_builtin(__builtin_amdgcn_s_wait_asynccnt)
  __builtin_amdgcn_s_wait_asynccnt(0);
#else
  asm volatile("s_wait_asynccnt 0" ::: "memory");
#endif
}

// ---------------- fragment loaders (ISA VGPR layouts, wave32) ----------------
// A 16x32 f16: lanes 0-15 K {0..7,16..23}, lanes 16-31 K {8..15,24..31}; row = per-lane M
__device__ __forceinline__ v16h load_A_frag(const _Float16* __restrict__ A, int row, int ks, int lane) {
  const _Float16* src = A + row * 64 + ks * 32 + ((lane & 16) ? 8 : 0);
  v8h lo = *(const v8h*)(src);        // K = kbase .. kbase+7
  v8h hi = *(const v8h*)(src + 16);   // K = kbase+16 .. kbase+23
  return __builtin_shufflevector(lo, hi, 0,1,2,3,4,5,6,7,8,9,10,11,12,13,14,15);
}
// B 32x16 f16 stored transposed as Bt[n][k] (Ktot = 64): lanes 0-15 K {0..7,16..23}, lanes 16-31 K {8..15,24..31}
__device__ __forceinline__ v16h load_B_frag(const _Float16* __restrict__ Bt, int col, int ks, int lane) {
  const _Float16* src = Bt + col * 64 + ks * 32 + ((lane & 16) ? 8 : 0);
  v8h lo = *(const v8h*)(src);
  v8h hi = *(const v8h*)(src + 16);
  return __builtin_shufflevector(lo, hi, 0,1,2,3,4,5,6,7,8,9,10,11,12,13,14,15);
}

// One 64->64 GEMM stage with bias+ReLU, A:[336][64] f16 LDS, Bt:[64][64] f16 LDS
// wave_u must be wave-uniform (readfirstlane'd) so loop control stays scalar.
__device__ __forceinline__ void gemm_stage(const _Float16* __restrict__ A,
                                           const _Float16* __restrict__ Bt,
                                           _Float16* __restrict__ Hout,
                                           const float* __restrict__ bias,
                                           int wave_u, int lane) {
  const int ntile = wave_u & 3;
  const int col   = ntile * 16 + (lane & 15);
  const float bv  = bias[col];
  const int rbase = (lane & 16) ? 8 : 0;
  for (int mt = (wave_u >> 2); mt < MT; mt += 2) {
    v8f acc = {};
    const int arow = mt * 16 + (lane & 15);
#pragma unroll
    for (int ks = 0; ks < 2; ++ks) {
      v16h a = load_A_frag(A, arow, ks, lane);
      v16h b = load_B_frag(Bt, col, ks, lane);
      acc = __builtin_amdgcn_wmma_f32_16x16x32_f16(false, a, false, b, (short)0, acc, false, false);
    }
#pragma unroll
    for (int v = 0; v < 8; ++v) {
      float x = acc[v] + bv;
      x = x > 0.0f ? x : 0.0f;
      Hout[(mt * 16 + rbase + v) * 64 + col] = (_Float16)x;
    }
  }
}

// ---------------- weight packing (fp32 -> f16 WMMA layouts) ----------------
__global__ void pack_weights(const float* __restrict__ w0, const float* __restrict__ w1,
                             const float* __restrict__ w2, const float* __restrict__ w3,
                             const float* __restrict__ w4, const float* __restrict__ wz,
                             const float* __restrict__ logs,
                             _Float16* __restrict__ wsB, float* __restrict__ scale16) {
  int i = blockIdx.x * blockDim.x + threadIdx.x;
  int stride = gridDim.x * blockDim.x;
  for (int idx = i; idx < W_TOTAL_H; idx += stride) {
    float v;
    if (idx < W_B1_OFF) {                       // Bt0[n][k], k = c*9+ky*3+kx, pad 54..63
      int n = idx >> 6, k = idx & 63;
      v = (k < 54) ? w0[n * 54 + k] : 0.0f;
    } else if (idx < W_BZ_OFF) {                // Bt1..Bt4: identical to OIHW flat for 1x1
      int r = idx - W_B1_OFF;
      int s = r >> 12;                          // which of w1..w4
      int rr = r & 4095;
      const float* w = (s == 0) ? w1 : (s == 1) ? w2 : (s == 2) ? w3 : w4;
      v = w[rr];                                // n*64 + k already
    } else {                                    // Bz2[rs][n][c] = wz[n][c*9 + rs]
      int r = idx - W_BZ_OFF;
      int rs = r >> 10;
      int rem = r & 1023;
      int n = rem >> 6, c = rem & 63;
      v = wz[n * 576 + c * 9 + rs];
    }
    wsB[idx] = (_Float16)v;
  }
  if (i < 16) scale16[i] = __expf(3.0f * logs[i]);
}

__global__ void init_logdet(const float* __restrict__ ld_in, float* __restrict__ ld_out) {
  int i = threadIdx.x;
  if (i < B_IMG) ld_out[i] = ld_in[i];
}

// ---------------- fused tile kernel ----------------
__global__ __launch_bounds__(256)
void flow_tile_kernel(const float* __restrict__ input, const float* __restrict__ ft,
                      const float* __restrict__ b0, const float* __restrict__ b1,
                      const float* __restrict__ b2, const float* __restrict__ b3,
                      const float* __restrict__ b4, const float* __restrict__ bz,
                      const _Float16* __restrict__ wsB, const float* __restrict__ scale16,
                      float* __restrict__ out, float* __restrict__ logdet) {
  extern __shared__ char smem[];
  _Float16* sW    = (_Float16*)smem;                 // 29696 halfs, at LDS offset 0
  _Float16* sHA   = sW + W_TOTAL_H;                  // 21504 halfs
  _Float16* sHB   = sHA + MROWS * 64;                // 21504 halfs (also A0 im2col)
  float*    sFeat = (float*)(sHB + MROWS * 64);      // 256*16 f32
  _Float16* sIn   = (_Float16*)sFeat;                // 2400 halfs, aliased (used only early)
  float*    sRed  = sFeat + 256 * 16;                // 256 f32

  const int t = threadIdx.x;
  const int lane = t & 31;
  const int wave_u = __builtin_amdgcn_readfirstlane(t >> 5);  // wave-uniform
  const int bx = blockIdx.x, by = blockIdx.y, b = blockIdx.z;
  const size_t imgbase = (size_t)b * IMG_STRIDE;

  // 1) weights -> LDS via CDNA5 async global->LDS b128 (bypasses VGPRs; ASYNCcnt)
  for (int i = t; i < W_TOTAL_B128; i += 256)
    async_g2l_b128((unsigned int)(i * 16), (const char*)wsB + i * 16);

  // 2) input halo tile: sIn[c][iy][ix], c: 0..3 = z1, 4..5 = ft; zero-padded
  for (int i = t; i < 6 * IT * IT; i += 256) {
    int c = i / (IT * IT);
    int rem = i - c * (IT * IT);
    int iy = rem / IT, ix = rem % IT;
    int gy = by * TILE - 2 + iy, gx = bx * TILE - 2 + ix;
    float v = 0.0f;
    if (gy >= 0 && gy < H_IMG && gx >= 0 && gx < W_IMG) {
      v = (c < 4) ? input[imgbase + (size_t)c * PLANE + gy * W_IMG + gx]
                  : ft[((size_t)b * 2 + (c - 4)) * PLANE + gy * W_IMG + gx];
    }
    sIn[i] = (_Float16)v;
  }
  wait_asynccnt0();                     // this wave's async LDS writes landed
  __syncthreads();
  // 3) im2col A0 into sHB: [336][64], K = c*9 + r*3 + s (54 valid), pad rows/K zero
  for (int i = t; i < MROWS * 64; i += 256) {
    int p = i >> 6, k = i & 63;
    _Float16 v = (_Float16)0.0f;
    if (p < NP && k < 54) {
      int py = p / HT, px = p - py * HT;
      int c = k / 9, rs = k - 9 * c, r = rs / 3, s = rs - 3 * r;
      v = sIn[c * (IT * IT) + (py + r) * IT + (px + s)];
    }
    sHB[i] = v;
  }
  __syncthreads();

  // 4) conv0 + conv1..4 (WMMA, f32 accumulate, bias+ReLU)
  gemm_stage(sHB, sW + W_B0_OFF,         sHA, b0, wave_u, lane); __syncthreads();
  gemm_stage(sHA, sW + W_B1_OFF,         sHB, b1, wave_u, lane); __syncthreads();
  gemm_stage(sHB, sW + W_B1_OFF + 4096,  sHA, b2, wave_u, lane); __syncthreads();
  gemm_stage(sHA, sW + W_B1_OFF + 8192,  sHB, b3, wave_u, lane); __syncthreads();
  gemm_stage(sHB, sW + W_B1_OFF + 12288, sHA, b4, wave_u, lane); __syncthreads();

  // 5) convz: 9 shifted K=64 GEMMs over h4 (= sHA), N = 16, then bias + exp(3*logs) scale
  {
    const _Float16* h4 = sHA;
    const _Float16* Bz = sW + W_BZ_OFF;
    const int col = lane & 15;
    const float bv = bz[col];
    const float sc = scale16[col];
    const int rbase = (lane & 16) ? 8 : 0;
    for (int mt = wave_u; mt < 16; mt += 8) {   // oy = mt, ox = lane&15
      v8f acc = {};
#pragma unroll
      for (int rs = 0; rs < 9; ++rs) {
        int r = rs / 3, s = rs - 3 * r;
        int row = (mt + r) * HT + col + s;
        const _Float16* Brs = Bz + rs * 1024;
#pragma unroll
        for (int ks = 0; ks < 2; ++ks) {
          v16h a = load_A_frag(h4, row, ks, lane);
          v16h bb = load_B_frag(Brs, col, ks, lane);
          acc = __builtin_amdgcn_wmma_f32_16x16x32_f16(false, a, false, bb, (short)0, acc, false, false);
        }
      }
#pragma unroll
      for (int v = 0; v < 8; ++v)
        sFeat[(mt * 16 + rbase + v) * 16 + col] = (acc[v] + bv) * sc;
    }
  }
  __syncthreads();

  // 6) RQ spline (2 bins) + shift + z1 passthrough + log-det reduction
  {
    const int oy = t >> 4, ox = t & 15;
    const int gy = by * TILE + oy, gx = bx * TILE + ox;
    const int pix = gy * W_IMG + gx;
    float ladsum = 0.0f;
#pragma unroll
    for (int c = 0; c < 4; ++c) {
      // passthrough z1
      out[imgbase + (size_t)c * PLANE + pix] = input[imgbase + (size_t)c * PLANE + pix];

      float x  = input[imgbase + (size_t)(4 + c) * PLANE + pix];
      float wv = sFeat[t * 16 + c * 4 + 0];
      float hv = sFeat[t * 16 + c * 4 + 1];
      float dv = sFeat[t * 16 + c * 4 + 2];
      float sh = sFeat[t * 16 + c * 4 + 3];

      bool inside = (x > -0.5f) && (x < 0.5f);
      float xs = inside ? x : 0.0f;
      float width = 1.0f / (1.0f + expf(-wv)) * 0.998f + 0.001f;
      float hh    = 1.0f / (1.0f + expf(-hv)) * 0.998f + 0.001f;
      float dd    = expf(dv) * 0.999f + 0.001f;
      float cw1 = width - 0.5f;                 // interior x-knot
      float ch1 = hh - 0.5f;                    // interior y-knot
      bool hi = (xs >= cw1);                    // bin index
      float in_cw = hi ? cw1 : -0.5f;
      float in_w  = hi ? (1.0f - width) : width;
      float in_ch = hi ? ch1 : -0.5f;
      float in_h  = hi ? (1.0f - hh) : hh;
      float in_delta = in_h / in_w;
      float d0 = hi ? dd : 1.0f;
      float d1 = hi ? 1.0f : dd;
      float theta = (xs - in_cw) / in_w;
      float omt = 1.0f - theta;
      float t1mt = theta * omt;
      float denom = in_delta + (d0 + d1 - 2.0f * in_delta) * t1mt;
      float o = in_ch + in_h * (in_delta * theta * theta + d0 * t1mt) / denom;
      float dnum = in_delta * in_delta * (d1 * theta * theta + 2.0f * in_delta * t1mt + d0 * omt * omt);
      float lad = logf(dnum) - 2.0f * logf(denom);
      o = fminf(fmaxf(o, -0.5f), 0.5f);
      o = inside ? o : x;
      lad = inside ? lad : 0.0f;
      out[imgbase + (size_t)(4 + c) * PLANE + pix] = o + sh;
      ladsum += lad;
    }
    sRed[t] = ladsum;
  }
  __syncthreads();
  for (int s = 128; s > 0; s >>= 1) {
    if (t < s) sRed[t] += sRed[t + s];
    __syncthreads();
  }
  if (t == 0) atomicAdd(&logdet[b], sRed[0]);
}

// ---------------- launch ----------------
extern "C" void kernel_launch(void* const* d_in, const int* in_sizes, int n_in,
                              void* d_out, int out_size, void* d_ws, size_t ws_size,
                              hipStream_t stream) {
  (void)in_sizes; (void)n_in; (void)out_size; (void)ws_size;
  const float* input     = (const float*)d_in[0];
  const float* logdet_in = (const float*)d_in[1];
  const float* ft        = (const float*)d_in[2];
  const float* w0 = (const float*)d_in[3];
  const float* b0 = (const float*)d_in[4];
  const float* w1 = (const float*)d_in[5];
  const float* b1 = (const float*)d_in[6];
  const float* w2 = (const float*)d_in[7];
  const float* b2 = (const float*)d_in[8];
  const float* w3 = (const float*)d_in[9];
  const float* b3 = (const float*)d_in[10];
  const float* w4 = (const float*)d_in[11];
  const float* b4 = (const float*)d_in[12];
  const float* wz = (const float*)d_in[13];
  const float* bz = (const float*)d_in[14];
  const float* logs = (const float*)d_in[15];

  _Float16* wsB   = (_Float16*)d_ws;
  float* scale16  = (float*)((char*)d_ws + W_TOTAL_B);

  float* out = (float*)d_out;
  float* logdet_out = out + (size_t)B_IMG * IMG_STRIDE;

  pack_weights<<<32, 256, 0, stream>>>(w0, w1, w2, w3, w4, wz, logs, wsB, scale16);
  init_logdet<<<1, 32, 0, stream>>>(logdet_in, logdet_out);
  dim3 grid(W_IMG / TILE, H_IMG / TILE, B_IMG);
  flow_tile_kernel<<<grid, 256, LDS_BYTES, stream>>>(input, ft, b0, b1, b2, b3, b4, bz,
                                                     wsB, scale16, out, logdet_out);
}
